// SAP_Block_89988154786303
// MI455X (gfx1250) — compile-verified, matrix operand order checked by reading, not run on hardware
//
#include <hip/hip_runtime.h>
#include <hip/hip_bf16.h>

typedef __attribute__((ext_vector_type(16))) _Float16 v16h;
typedef __attribute__((ext_vector_type(8)))  _Float16 v8h;
typedef __attribute__((ext_vector_type(8)))  float    v8f;

#define AS3 __attribute__((address_space(3)))
typedef AS3 _Float16 lf16;
typedef AS3 v8h      lv8h;

#define C_DIM 512
#define HID   1024
#define SEQ   1536
#define BATCH 4
#define NTOK  (BATCH*SEQ)      // 6144
#define NHEAD 8
#define HD    64
#define SSEG  512              // SEQ/3
#define ALPHA 0.5f

// GEMM block tiling
#define BM 128                 // block M rows (4 waves x 32)
#define BN 64                  // block N cols
#define BK 64                  // K chunk staged per buffer
#define PAD 8                  // halves of padding per LDS row
#define LDSTR (BK + PAD)       // 72 halves -> 144B row stride (16B aligned)

// ---------------- workspace layout (bytes) ----------------
#define H16_OFF   0u           // 6144*512 f16   = 6291456
#define QKVT_OFF  6291456u     // 1536*512 f16   = 1572864
#define Q_OFF     7864320u     // 4*8*1536*64 f16= 6291456
#define K_OFF     14155776u
#define V_OFF     20447232u    // Vt [B,H,d,SEQ]
#define X2_OFF    26738688u    // 6144*512 f32   = 12582912
#define H2_OFF    39321600u    // 6144*512 f16
#define FC1T_OFF  45613056u    // 1024*512 f16   = 1048576
#define FC2T_OFF  46661632u    // 512*1024 f16   = 1048576
#define M16_OFF   47710208u    // 6144*1024 f16  = 12582912  -> end 60293120

// Generic LDS pointer -> addrspace(3) pointer. Flat LDS pointers carry the
// wave-relative LDS byte offset in their low 32 bits; AS3 pointers ARE that
// 32-bit offset.
__device__ __forceinline__ lf16* to_lds(_Float16* p) {
  return (lf16*)(unsigned int)(uintptr_t)p;
}

// ---------------- async copy: 16B global -> LDS (ASYNCcnt-tracked) ----------
__device__ __forceinline__ void async_cp16(const _Float16* g, lf16* l) {
  unsigned int loff = (unsigned int)(uintptr_t)l;   // LDS byte offset
  unsigned long long ga = (unsigned long long)(uintptr_t)g;
  asm volatile("global_load_async_to_lds_b128 %0, %1, off"
               :: "v"(loff), "v"(ga) : "memory");
}
__device__ __forceinline__ void wait_async12() {
  asm volatile("s_wait_asynccnt 0xc" ::: "memory");
}
__device__ __forceinline__ void wait_async0() {
  asm volatile("s_wait_asynccnt 0x0" ::: "memory");
}

// ---------------- fragment loaders ----------------
// 16x32 f16 tile, rows contiguous at `stride` halves, into WMMA A/B layout:
// lane = hi*16 + m ; elems 0..7 -> k = 8*hi+0..7 ; 8..15 -> k = 16+8*hi+0..7
__device__ __forceinline__ v16h load_frag(const _Float16* p, int stride) {
  int lane = threadIdx.x & 31;
  const _Float16* q = p + (size_t)(lane & 15) * stride + ((lane >> 4) << 3);
  v8h a = *(const v8h*)q;
  v8h b = *(const v8h*)(q + 16);
  v16h r;
#pragma unroll
  for (int i = 0; i < 8; ++i) { r[i] = a[i]; r[i + 8] = b[i]; }
  return r;
}
// Same, but guaranteed ds_load_b128 via addrspace(3) pointers.
__device__ __forceinline__ v16h load_frag_lds(const lf16* p, int stride) {
  int lane = threadIdx.x & 31;
  const lf16* q = p + (lane & 15) * stride + ((lane >> 4) << 3);
  v8h a = *(const lv8h*)q;
  v8h b = *(const lv8h*)(q + 16);
  v16h r;
#pragma unroll
  for (int i = 0; i < 8; ++i) { r[i] = a[i]; r[i + 8] = b[i]; }
  return r;
}

// ---------------- cooperative staging of one K-chunk into LDS ---------------
// A chunk: BM x BK (8 issues/thread), B chunk: BN x BK (4 issues/thread).
__device__ __forceinline__ void stage_chunk(const _Float16* __restrict__ Ablk, int lda,
                                            const _Float16* __restrict__ Bblk, int ldb,
                                            int kc, lf16* Abuf, lf16* Bbuf) {
  int tid = threadIdx.x;
#pragma unroll
  for (int j = 0; j < 8; ++j) {
    int idx = j * 128 + tid;          // 0..1023 units of 8 halves
    int row = idx >> 3;               // 0..127
    int col = (idx & 7) * 8;          // 0..56
    async_cp16(Ablk + (size_t)row * lda + kc * BK + col, Abuf + row * LDSTR + col);
  }
#pragma unroll
  for (int j = 0; j < 4; ++j) {
    int idx = j * 128 + tid;          // 0..511
    int row = idx >> 3;               // 0..63
    int col = (idx & 7) * 8;
    async_cp16(Bblk + (size_t)row * ldb + kc * BK + col, Bbuf + row * LDSTR + col);
  }
}

// ---------------- staged block GEMM core: 128x64 per block, 32x64 per wave ---
// BT is B transposed ([N][K] row-major). acc[set][t] holds rows w*32+set*16..+16.
__device__ __forceinline__ void gemm_block(const _Float16* __restrict__ Ablk, int lda,
                                           const _Float16* __restrict__ Bblk, int ldb,
                                           int K, lf16* As0, lf16* As1,
                                           lf16* Bs0, lf16* Bs1,
                                           v8f acc[2][4]) {
  int w = threadIdx.x >> 5;
  int NC = K / BK;
  stage_chunk(Ablk, lda, Bblk, ldb, 0, As0, Bs0);
  stage_chunk(Ablk, lda, Bblk, ldb, 1, As1, Bs1);
  for (int kc = 0; kc < NC; ++kc) {
    if (kc + 1 < NC) wait_async12();   // our chunk kc retired (in-order ASYNCcnt)
    else             wait_async0();
    __syncthreads();                   // all waves' chunk-kc data visible in LDS
    const lf16* Ab = ((kc & 1) ? As1 : As0) + (w * 32) * LDSTR;
    const lf16* Bb = (kc & 1) ? Bs1 : Bs0;
#pragma unroll
    for (int ks = 0; ks < 2; ++ks) {
      v16h a0 = load_frag_lds(Ab + ks * 32, LDSTR);
      v16h a1 = load_frag_lds(Ab + 16 * LDSTR + ks * 32, LDSTR);
#pragma unroll
      for (int t = 0; t < 4; ++t) {
        v16h bf = load_frag_lds(Bb + (t * 16) * LDSTR + ks * 32, LDSTR);
        acc[0][t] = __builtin_amdgcn_wmma_f32_16x16x32_f16(
            false, a0, false, bf, (short)0, acc[0][t], false, false);
        acc[1][t] = __builtin_amdgcn_wmma_f32_16x16x32_f16(
            false, a1, false, bf, (short)0, acc[1][t], false, false);
      }
    }
    __syncthreads();                   // everyone done reading buf[kc&1]
    if (kc + 2 < NC)
      stage_chunk(Ablk, lda, Bblk, ldb, kc + 2,
                  (kc & 1) ? As1 : As0, (kc & 1) ? Bs1 : Bs0);
  }
}

// ---------------- LayerNorm (rowwise, 512 cols) -> f16 ----------------
__global__ __launch_bounds__(128) void ln_kernel(const float* __restrict__ x,
                                                 const float* __restrict__ g,
                                                 const float* __restrict__ b,
                                                 _Float16* __restrict__ out) {
  int row = blockIdx.x, tid = threadIdx.x;
  float4 v = ((const float4*)(x + (size_t)row * C_DIM))[tid];
  float s  = v.x + v.y + v.z + v.w;
  float sq = v.x * v.x + v.y * v.y + v.z * v.z + v.w * v.w;
#pragma unroll
  for (int off = 16; off; off >>= 1) {
    s  += __shfl_xor(s,  off);
    sq += __shfl_xor(sq, off);
  }
  __shared__ float rs[4], rq[4];
  if ((tid & 31) == 0) { rs[tid >> 5] = s; rq[tid >> 5] = sq; }
  __syncthreads();
  float ts = rs[0] + rs[1] + rs[2] + rs[3];
  float tq = rq[0] + rq[1] + rq[2] + rq[3];
  float mean = ts * (1.0f / C_DIM);
  float var  = tq * (1.0f / C_DIM) - mean * mean;
  float rstd = rsqrtf(var + 1e-5f);
  float4 gv = ((const float4*)g)[tid];
  float4 bv = ((const float4*)b)[tid];
  _Float16* o = out + (size_t)row * C_DIM + tid * 4;
  o[0] = (_Float16)((v.x - mean) * rstd * gv.x + bv.x);
  o[1] = (_Float16)((v.y - mean) * rstd * gv.y + bv.y);
  o[2] = (_Float16)((v.z - mean) * rstd * gv.z + bv.z);
  o[3] = (_Float16)((v.w - mean) * rstd * gv.w + bv.w);
}

// ---------------- weight convert + transpose: w[K][N] f32 -> wt[N][K] f16 ----
__global__ __launch_bounds__(256) void cvt_t_kernel(const float* __restrict__ w,
                                                    _Float16* __restrict__ wt,
                                                    int K, int N) {
  int i = blockIdx.x * 256 + threadIdx.x;
  int k = i / N, n = i % N;
  wt[(size_t)n * K + k] = (_Float16)w[(size_t)k * N + n];
}

// ---------------- QKV GEMM (staged): h16 @ qkvT -> Q16(scaled)/K16/Vt16 ------
__global__ __launch_bounds__(128) void qkv_kernel(const _Float16* __restrict__ h16,
                                                  const _Float16* __restrict__ wT,
                                                  const float* __restrict__ bias,
                                                  _Float16* __restrict__ Q,
                                                  _Float16* __restrict__ Kk,
                                                  _Float16* __restrict__ Vt) {
  __shared__ _Float16 As[2][BM * LDSTR];
  __shared__ _Float16 Bs[2][BN * LDSTR];
  int bm = blockIdx.x / 24, bn = blockIdx.x % 24;
  v8f acc[2][4] = {};
  gemm_block(h16 + (size_t)bm * BM * C_DIM, C_DIM,
             wT + (size_t)bn * BN * C_DIM, C_DIM, C_DIM,
             to_lds(As[0]), to_lds(As[1]), to_lds(Bs[0]), to_lds(Bs[1]), acc);
  int lane = threadIdx.x & 31, l15 = lane & 15, hi = lane >> 4, w = threadIdx.x >> 5;
  int seg = bn >> 3, h = bn & 7;
#pragma unroll
  for (int set = 0; set < 2; ++set) {
    int rowbase = bm * BM + w * 32 + set * 16;
#pragma unroll
    for (int t = 0; t < 4; ++t) {
      int dd  = t * 16 + l15;
      float bc = bias[bn * 64 + dd];
#pragma unroll
      for (int r = 0; r < 8; ++r) {
        int mrow = rowbase + r + 8 * hi;
        int b = mrow / SEQ, tok = mrow % SEQ;
        float v = acc[set][t][r] + bc;
        size_t bh = (size_t)(b * NHEAD + h);
        if (seg == 0)      Q [(bh * SEQ + tok) * HD + dd] = (_Float16)(v * 0.125f);
        else if (seg == 1) Kk[(bh * SEQ + tok) * HD + dd] = (_Float16)v;
        else               Vt[(bh * HD + dd) * SEQ + tok] = (_Float16)v;
      }
    }
  }
}

// ---------------- attention: one wave = (b,h, query-triple of 16 rows) -------
__global__ __launch_bounds__(256, 1) void attn_kernel(const _Float16* __restrict__ Q,
                                                      const _Float16* __restrict__ Km,
                                                      const _Float16* __restrict__ Vt,
                                                      const float* __restrict__ x,
                                                      float* __restrict__ x2) {
  __shared__ _Float16 lds_p[8][16 * 32];
  int wid = threadIdx.x >> 5;
  int lane = threadIdx.x & 31, l15 = lane & 15, hi = lane >> 4;
  int W = blockIdx.x * 8 + wid;
  int r16 = W & 31;
  int bh = W >> 5;                      // 0..31
  const _Float16* Qb = Q  + (size_t)bh * SEQ * HD;
  const _Float16* Kb = Km + (size_t)bh * SEQ * HD;
  const _Float16* Vb = Vt + (size_t)bh * HD * SEQ;

  v16h aq[3][2];
#pragma unroll
  for (int s = 0; s < 3; ++s) {
    int qr = s * SSEG + r16 * 16;
#pragma unroll
    for (int kk = 0; kk < 2; ++kk)
      aq[s][kk] = load_frag(Qb + (size_t)qr * HD + kk * 32, HD);
  }

  float mr[3][8], lr[3][8];
  v8f oacc[3][4] = {};
#pragma unroll
  for (int s = 0; s < 3; ++s)
#pragma unroll
    for (int r = 0; r < 8; ++r) { mr[s][r] = -1e30f; lr[s][r] = 0.0f; }

  lf16* myp = to_lds(&lds_p[wid][0]);

  for (int kc = 0; kc < 48; ++kc) {   // 48 chunks of 32 keys
    v16h bk[2][2];
#pragma unroll
    for (int t = 0; t < 2; ++t)
#pragma unroll
      for (int kk = 0; kk < 2; ++kk)
        bk[t][kk] = load_frag(Kb + (size_t)(kc * 32 + t * 16) * HD + kk * 32, HD);
    v16h bv[4];
#pragma unroll
    for (int dt = 0; dt < 4; ++dt)
      bv[dt] = load_frag(Vb + (size_t)(dt * 16) * SEQ + kc * 32, SEQ);

    v8f sc[3][2];
    v8f zc = {};
#pragma unroll
    for (int s = 0; s < 3; ++s)
#pragma unroll
      for (int t = 0; t < 2; ++t) {
        v8f tmp = __builtin_amdgcn_wmma_f32_16x16x32_f16(
            false, aq[s][0], false, bk[t][0], (short)0, zc, false, false);
        sc[s][t] = __builtin_amdgcn_wmma_f32_16x16x32_f16(
            false, aq[s][1], false, bk[t][1], (short)0, tmp, false, false);
      }
    // segment coupling (exact per-column): s1 += a*s0 ; s2 += a*s1
#pragma unroll
    for (int t = 0; t < 2; ++t) {
      sc[1][t] = sc[1][t] + ALPHA * sc[0][t];
      sc[2][t] = sc[2][t] + ALPHA * sc[1][t];
    }

#pragma unroll
    for (int s = 0; s < 3; ++s) {
#pragma unroll
      for (int r = 0; r < 8; ++r) {
        float x0 = sc[s][0][r], x1 = sc[s][1][r];
        float mv = fmaxf(x0, x1);
#pragma unroll
        for (int off = 8; off; off >>= 1) mv = fmaxf(mv, __shfl_xor(mv, off));
        float mn  = fmaxf(mr[s][r], mv);
        float rscl = __expf(mr[s][r] - mn);
        float p0 = __expf(x0 - mn);
        float p1 = __expf(x1 - mn);
        float rsum = p0 + p1;
#pragma unroll
        for (int off = 8; off; off >>= 1) rsum += __shfl_xor(rsum, off);
        lr[s][r] = lr[s][r] * rscl + rsum;
        mr[s][r] = mn;
#pragma unroll
        for (int dt = 0; dt < 4; ++dt) oacc[s][dt][r] *= rscl;
        int row = r + 8 * hi;
        myp[row * 32 + l15]      = (_Float16)p0;
        myp[row * 32 + 16 + l15] = (_Float16)p1;
      }
      asm volatile("s_wait_dscnt 0x0" ::: "memory");
      // reload P tile (16x32) as an A fragment from LDS (ds_load_b128)
      v16h ap = load_frag_lds(myp, 32);
#pragma unroll
      for (int dt = 0; dt < 4; ++dt)
        oacc[s][dt] = __builtin_amdgcn_wmma_f32_16x16x32_f16(
            false, ap, false, bv[dt], (short)0, oacc[s][dt], false, false);
    }
  }

  // epilogue: normalize, fuse residual  x2 = x + o
  int h = bh & 7, b = bh >> 3;
#pragma unroll
  for (int s = 0; s < 3; ++s)
#pragma unroll
    for (int r = 0; r < 8; ++r) {
      float inv = 1.0f / lr[s][r];
      int tok = s * SSEG + r16 * 16 + r + 8 * hi;
#pragma unroll
      for (int dt = 0; dt < 4; ++dt) {
        int col = h * HD + dt * 16 + l15;
        size_t idx = ((size_t)b * SEQ + tok) * C_DIM + col;
        x2[idx] = x[idx] + oacc[s][dt][r] * inv;
      }
    }
}

// ---------------- FC1 (staged) + exact GELU -> m16 f16 ----------------
__global__ __launch_bounds__(128) void fc1_kernel(const _Float16* __restrict__ h2,
                                                  const _Float16* __restrict__ wT,
                                                  const float* __restrict__ bias,
                                                  _Float16* __restrict__ m16) {
  __shared__ _Float16 As[2][BM * LDSTR];
  __shared__ _Float16 Bs[2][BN * LDSTR];
  int bm = blockIdx.x / 16, bn = blockIdx.x % 16;
  v8f acc[2][4] = {};
  gemm_block(h2 + (size_t)bm * BM * C_DIM, C_DIM,
             wT + (size_t)bn * BN * C_DIM, C_DIM, C_DIM,
             to_lds(As[0]), to_lds(As[1]), to_lds(Bs[0]), to_lds(Bs[1]), acc);
  int lane = threadIdx.x & 31, l15 = lane & 15, hi = lane >> 4, w = threadIdx.x >> 5;
#pragma unroll
  for (int set = 0; set < 2; ++set) {
    int rowbase = bm * BM + w * 32 + set * 16;
#pragma unroll
    for (int t = 0; t < 4; ++t) {
      int col = bn * 64 + t * 16 + l15;
      float bc = bias[col];
#pragma unroll
      for (int r = 0; r < 8; ++r) {
        int row = rowbase + r + 8 * hi;
        float v = acc[set][t][r] + bc;
        float g = 0.5f * v * (1.0f + erff(v * 0.70710678118654752f));
        m16[(size_t)row * HID + col] = (_Float16)g;
      }
    }
  }
}

// ---------------- FC2 (staged) + bias + residual -> out f32 ----------------
__global__ __launch_bounds__(128) void fc2_kernel(const _Float16* __restrict__ m16,
                                                  const _Float16* __restrict__ wT,
                                                  const float* __restrict__ bias,
                                                  const float* __restrict__ x2,
                                                  float* __restrict__ out) {
  __shared__ _Float16 As[2][BM * LDSTR];
  __shared__ _Float16 Bs[2][BN * LDSTR];
  int bm = blockIdx.x / 8, bn = blockIdx.x % 8;
  v8f acc[2][4] = {};
  gemm_block(m16 + (size_t)bm * BM * HID, HID,
             wT + (size_t)bn * BN * HID, HID, HID,
             to_lds(As[0]), to_lds(As[1]), to_lds(Bs[0]), to_lds(Bs[1]), acc);
  int lane = threadIdx.x & 31, l15 = lane & 15, hi = lane >> 4, w = threadIdx.x >> 5;
#pragma unroll
  for (int set = 0; set < 2; ++set) {
    int rowbase = bm * BM + w * 32 + set * 16;
#pragma unroll
    for (int t = 0; t < 4; ++t) {
      int col = bn * 64 + t * 16 + l15;
      float bc = bias[col];
#pragma unroll
      for (int r = 0; r < 8; ++r) {
        int row = rowbase + r + 8 * hi;
        size_t idx = (size_t)row * C_DIM + col;
        out[idx] = x2[idx] + acc[set][t][r] + bc;
      }
    }
  }
}

extern "C" void kernel_launch(void* const* d_in, const int* in_sizes, int n_in,
                              void* d_out, int out_size, void* d_ws, size_t ws_size,
                              hipStream_t stream) {
  const float* x      = (const float*)d_in[0];
  const float* ln1_g  = (const float*)d_in[1];
  const float* ln1_b  = (const float*)d_in[2];
  const float* qkv_w  = (const float*)d_in[3];
  const float* qkv_b  = (const float*)d_in[4];
  const float* ln2_g  = (const float*)d_in[5];
  const float* ln2_b  = (const float*)d_in[6];
  const float* fc1_w  = (const float*)d_in[7];
  const float* fc1_b  = (const float*)d_in[8];
  const float* fc2_w  = (const float*)d_in[9];
  const float* fc2_b  = (const float*)d_in[10];
  float* out = (float*)d_out;

  char* ws = (char*)d_ws;
  _Float16* h16   = (_Float16*)(ws + H16_OFF);
  _Float16* qkvT  = (_Float16*)(ws + QKVT_OFF);
  _Float16* Q16   = (_Float16*)(ws + Q_OFF);
  _Float16* K16   = (_Float16*)(ws + K_OFF);
  _Float16* Vt16  = (_Float16*)(ws + V_OFF);
  float*    x2    = (float*)   (ws + X2_OFF);
  _Float16* h2    = (_Float16*)(ws + H2_OFF);
  _Float16* fc1T  = (_Float16*)(ws + FC1T_OFF);
  _Float16* fc2T  = (_Float16*)(ws + FC2T_OFF);
  _Float16* m16   = (_Float16*)(ws + M16_OFF);

  // LN1 -> f16
  ln_kernel<<<NTOK, 128, 0, stream>>>(x, ln1_g, ln1_b, h16);
  // weight convert+transpose (f32 [K][N] -> f16 [N][K])
  cvt_t_kernel<<<(C_DIM * 3 * C_DIM) / 256, 256, 0, stream>>>(qkv_w, qkvT, C_DIM, 3 * C_DIM);
  cvt_t_kernel<<<(C_DIM * HID) / 256, 256, 0, stream>>>(fc1_w, fc1T, C_DIM, HID);
  cvt_t_kernel<<<(HID * C_DIM) / 256, 256, 0, stream>>>(fc2_w, fc2T, HID, C_DIM);
  // QKV projection (scale folded into Q, V stored transposed)
  qkv_kernel<<<(NTOK / BM) * 24, 128, 0, stream>>>(h16, qkvT, qkv_b, Q16, K16, Vt16);
  // segment-coupled flash attention + residual
  attn_kernel<<<BATCH * NHEAD * (SSEG / 16) / 8, 256, 0, stream>>>(Q16, K16, Vt16, x, x2);
  // LN2 -> f16
  ln_kernel<<<NTOK, 128, 0, stream>>>(x2, ln2_g, ln2_b, h2);
  // MLP
  fc1_kernel<<<(NTOK / BM) * 16, 128, 0, stream>>>(h2, fc1T, fc1_b, m16);
  fc2_kernel<<<(NTOK / BM) * 8, 128, 0, stream>>>(m16, fc2T, fc2_b, x2, out);
}
